// TS_EGNN_64527588655492
// MI455X (gfx1250) — matrix operand order repaired and checked by reading, not compiled, not run on hardware
//
#include <hip/hip_runtime.h>
#include <hip/hip_bf16.h>

// ---------------------------------------------------------------------------
// EGNN (N=30000, E=240000, H=64, 3 layers) for MI455X (gfx1250, wave32).
// All GEMMs run on v_wmma_f32_16x16x32_bf16. One wave = one 16-row tile.
// Edge-message MLP (203->406->16) fully fused per 16-edge tile: the [16,416]
// intermediate lives in LDS only. Weights pre-converted to zero-padded bf16
// (L2-resident). Scatters use global f32 atomics.
// Round-1 deltas: SiLU via v_rcp_f32 (no IEEE div expansion); bf16 converts
// via native fptrunc (hardware v_cvt bf16 path) instead of integer rounding.
// ---------------------------------------------------------------------------

#define NN     30000
#define EE     240000
#define HH     64
#define NGRAPH 32

typedef __attribute__((ext_vector_type(16))) __bf16          v16bf;
typedef __attribute__((ext_vector_type(8)))  float           v8f;
typedef __attribute__((ext_vector_type(8)))  unsigned short  v8u16;

union frag_u { v16bf b; v8u16 h[2]; };

__device__ __forceinline__ unsigned short f2bf(float f) {
  union { __bf16 b; unsigned short s; } u;
  u.b = (__bf16)f;                       // native fptrunc -> v_cvt*bf16*
  return u.s;
}
__device__ __forceinline__ float siluf(float x) {
  return x * __builtin_amdgcn_rcpf(1.0f + __expf(-x));   // trans ops only
}

// A fragment (16x32 bf16, M rows in LDS row-major [16][kpad]).
// ISA layout: lanes 0-15 -> row=lane, K {kc*32+0..7, +16..23};
//             lanes 16-31 -> row=lane-16, K {+8..15, +24..31}.
__device__ __forceinline__ v16bf load_a_frag(const unsigned short* base, int kpad, int kc) {
  int lane = threadIdx.x & 31;
  const unsigned short* p = base + (lane & 15) * kpad + kc * 32 + ((lane & 16) ? 8 : 0);
  frag_u f;
  f.h[0] = *(const v8u16*)(p);
  f.h[1] = *(const v8u16*)(p + 16);
  return f.b;
}
// B fragment (32x16 bf16) from global row-major padded weight [Kpad][Nout]:
// lane L holds column n = nt*16+(L&15), K {kc*32+8h+0..7, +16..23} (h = L/16).
__device__ __forceinline__ v16bf load_b_frag(const unsigned short* w, int nout, int kc, int nt) {
  int lane = threadIdx.x & 31;
  const unsigned short* p = w + (size_t)(kc * 32 + ((lane & 16) ? 8 : 0)) * nout + nt * 16 + (lane & 15);
  frag_u f;
#pragma unroll
  for (int s = 0; s < 8; ++s) {
    f.h[0][s] = p[(size_t)s * nout];
    f.h[1][s] = p[(size_t)(16 + s) * nout];
  }
  return f.b;
}
#define WMMA_BF16(a, b, c) \
  __builtin_amdgcn_wmma_f32_16x16x32_bf16(false, (a), false, (b), (short)0, (c), false, false)

// ---------------------------------------------------------------------------
// Weight convert + zero-pad: f32 [rows][cols] -> bf16 [rpad][cpad]
// ---------------------------------------------------------------------------
__global__ void convert_pad_kernel(const float* __restrict__ src, int rows, int cols,
                                   unsigned short* __restrict__ dst, int rpad, int cpad) {
  int i = blockIdx.x * blockDim.x + threadIdx.x;
  if (i >= rpad * cpad) return;
  int r = i / cpad, c = i - r * cpad;
  float v = (r < rows && c < cols) ? src[r * cols + c] : 0.0f;
  dst[i] = f2bf(v);
}

__global__ void f32_to_bf16_kernel(const float* __restrict__ src,
                                   unsigned short* __restrict__ dst, int n) {
  int i = blockIdx.x * blockDim.x + threadIdx.x;
  if (i < n) dst[i] = f2bf(src[i]);
}

// emask from ORIGINAL positions (radius graph, cutoff 3.0)
__global__ void emask_kernel(const float* __restrict__ pos, const int* __restrict__ ei0,
                             const int* __restrict__ ei1, float* __restrict__ emask, int E) {
  int e = blockIdx.x * blockDim.x + threadIdx.x;
  if (e >= E) return;
  int r = ei0[e], c = ei1[e];
  float dx = pos[r * 3 + 0] - pos[c * 3 + 0];
  float dy = pos[r * 3 + 1] - pos[c * 3 + 1];
  float dz = pos[r * 3 + 2] - pos[c * 3 + 2];
  emask[e] = (sqrtf(dx * dx + dy * dy + dz * dz) < 3.0f) ? 1.0f : 0.0f;
}

// ---------------------------------------------------------------------------
// Generic WMMA GEMM: out = [act](A_f32[M][Kin] @ Bw_bf16[KPAD][Nout] + bias)
// One wave per 16-row tile; A converted to bf16 in LDS.
// ---------------------------------------------------------------------------
template <int KPAD, bool SILU_ACT, bool OUT_BF16>
__global__ void wmma_gemm_kernel(const float* __restrict__ A, int M, int Kin,
                                 const unsigned short* __restrict__ Bw, int Nout,
                                 const float* __restrict__ bias,
                                 float* __restrict__ outF, unsigned short* __restrict__ outH) {
  constexpr int WAVES = 4;
  __shared__ alignas(16) unsigned short astg[WAVES][16 * KPAD];
  int w = threadIdx.x >> 5, lane = threadIdx.x & 31;
  int m0 = (blockIdx.x * WAVES + w) * 16;
  if (m0 >= M) return;
  {
    int row = lane & 15;
    int cbeg = (lane & 16) ? (KPAD / 2) : 0;
    const float* src = A + (size_t)(m0 + row) * Kin;
    for (int c = cbeg; c < cbeg + KPAD / 2; ++c)
      astg[w][row * KPAD + c] = (c < Kin) ? f2bf(src[c]) : (unsigned short)0;
  }
  int col = lane & 15, rbase = (lane & 16) ? 8 : 0;
  __builtin_prefetch(Bw, 0, 1);
  for (int nt = 0; nt < Nout / 16; ++nt) {
    int gc = nt * 16 + col;
    float bv = bias[gc];
    v8f acc;
#pragma unroll
    for (int v = 0; v < 8; ++v) acc[v] = bv;
#pragma unroll
    for (int kc = 0; kc < KPAD / 32; ++kc)
      acc = WMMA_BF16(load_a_frag(&astg[w][0], KPAD, kc), load_b_frag(Bw, Nout, kc, nt), acc);
#pragma unroll
    for (int v = 0; v < 8; ++v) {
      float x = acc[v];
      if (SILU_ACT) x = siluf(x);
      size_t idx = (size_t)(m0 + rbase + v) * Nout + gc;
      if (OUT_BF16) outH[idx] = f2bf(x); else outF[idx] = x;
    }
  }
}

// ---------------------------------------------------------------------------
// Per-graph LayerNorm statistics: stats[g] = {count, sum, sumsq}
// ---------------------------------------------------------------------------
__global__ void graph_stats_kernel(const float* __restrict__ feats, const int* __restrict__ batch,
                                   float* __restrict__ stats, int N) {
  __shared__ float ls[NGRAPH * 3];
  for (int i = threadIdx.x; i < NGRAPH * 3; i += blockDim.x) ls[i] = 0.0f;
  __syncthreads();
  int n = blockIdx.x * blockDim.x + threadIdx.x;
  if (n < N) {
    const float* f = feats + (size_t)n * HH;
    float s = 0.0f, ss = 0.0f;
    for (int c = 0; c < HH; ++c) { float x = f[c]; s += x; ss += x * x; }
    int g = batch[n];
    atomicAdd(&ls[g * 3 + 0], 1.0f);
    atomicAdd(&ls[g * 3 + 1], s);
    atomicAdd(&ls[g * 3 + 2], ss);
  }
  __syncthreads();
  for (int i = threadIdx.x; i < NGRAPH * 3; i += blockDim.x)
    if (ls[i] != 0.0f) atomicAdd(&stats[i], ls[i]);
}

// ---------------------------------------------------------------------------
// Fused edge kernel: per 16-edge tile (1 wave):
//   build msg[16][224] bf16 (feats[col]|feats[row]|edge_feat|fourier, padded)
//   GEMM1 (K=224, N=416) -> SiLU -> LDS bf16
//   GEMM2 (K=416, N=16)  -> SiLU -> m
//   coors MLP (16->64 silu -> dot Wc2) -> w ; scatter coors delta + m_i
// ---------------------------------------------------------------------------
__global__ void edge_layer_kernel(
    const float* __restrict__ coors, const unsigned short* __restrict__ featsH,
    const unsigned short* __restrict__ edgeFeatH, const float* __restrict__ emask,
    const int* __restrict__ ei0, const int* __restrict__ ei1,
    const unsigned short* __restrict__ We1, const float* __restrict__ be1,
    const unsigned short* __restrict__ We2, const float* __restrict__ be2,
    const unsigned short* __restrict__ Wc1, const float* __restrict__ bc1,
    const float* __restrict__ Wc2, const float* __restrict__ bc2,
    const float* __restrict__ coors_scale,
    float* __restrict__ coorsNext, float* __restrict__ mAgg, int E) {
  constexpr int WAVES = 2;
  __shared__ alignas(16) unsigned short msg[WAVES][16 * 224];
  __shared__ alignas(16) unsigned short h1 [WAVES][16 * 416];
  __shared__ alignas(16) unsigned short mst[WAVES][16 * 32];
  __shared__ alignas(16) float          tst[WAVES][16 * 64];
  __shared__ int   cn[WAVES][16];
  __shared__ float em[WAVES][16];

  int w = threadIdx.x >> 5, lane = threadIdx.x & 31;
  int e0 = (blockIdx.x * WAVES + w) * 16;
  if (e0 >= E) return;

  float relx = 0.f, rely = 0.f, relz = 0.f, inv = 0.f;
  {
    int L = lane & 15;
    int e = e0 + L;
    int r = ei0[e];
    if (lane < 16) {
      int c = ei1[e];
      cn[w][L] = c;
      em[w][L] = emask[e];
      float rx = coors[(size_t)r * 3 + 0] - coors[(size_t)c * 3 + 0];
      float ry = coors[(size_t)r * 3 + 1] - coors[(size_t)c * 3 + 1];
      float rz = coors[(size_t)r * 3 + 2] - coors[(size_t)c * 3 + 2];
      relx = rx; rely = ry; relz = rz;
      float d2 = rx * rx + ry * ry + rz * rz;
      inv = 1.0f / fmaxf(sqrtf(d2), 1e-8f);
#pragma unroll
      for (int k = 0; k < 5; ++k) {
        float ds = d2 * (1.0f / (float)(1 << k));
        msg[w][L * 224 + 192 + k] = f2bf(__sinf(ds));
        msg[w][L * 224 + 197 + k] = f2bf(__cosf(ds));
      }
      msg[w][L * 224 + 202] = f2bf(d2);
      for (int c2 = 203; c2 < 224; ++c2) msg[w][L * 224 + c2] = 0;
      const v8u16* s = (const v8u16*)(featsH + (size_t)c * HH);     // feats[col] -> 0..63
      v8u16* d = (v8u16*)&msg[w][L * 224 + 0];
#pragma unroll
      for (int q = 0; q < 8; ++q) d[q] = s[q];
    } else {
      const v8u16* s = (const v8u16*)(featsH + (size_t)r * HH);     // feats[row] -> 64..127
      v8u16* d = (v8u16*)&msg[w][L * 224 + 64];
#pragma unroll
      for (int q = 0; q < 8; ++q) d[q] = s[q];
      s = (const v8u16*)(edgeFeatH + (size_t)e * HH);               // edge_feat -> 128..191
      d = (v8u16*)&msg[w][L * 224 + 128];
#pragma unroll
      for (int q = 0; q < 8; ++q) d[q] = s[q];
    }
  }
  int col = lane & 15, rbase = (lane & 16) ? 8 : 0;
  __builtin_prefetch(We1, 0, 1);
  __builtin_prefetch(We2, 0, 1);

  // ---- GEMM1: [16,224] x [224,416] + be1, SiLU -> h1 (bf16) ----
  for (int nt = 0; nt < 26; ++nt) {
    int gc = nt * 16 + col;
    float bv = (gc < 406) ? be1[gc] : 0.0f;
    v8f acc;
#pragma unroll
    for (int v = 0; v < 8; ++v) acc[v] = bv;
#pragma unroll
    for (int kc = 0; kc < 7; ++kc)
      acc = WMMA_BF16(load_a_frag(&msg[w][0], 224, kc), load_b_frag(We1, 416, kc, nt), acc);
#pragma unroll
    for (int v = 0; v < 8; ++v)
      h1[w][(rbase + v) * 416 + gc] = f2bf(siluf(acc[v]));
  }

  // ---- GEMM2: [16,416] x [416,16] + be2, SiLU -> m ----
  v8f mf;
  {
    float bv = be2[col];
#pragma unroll
    for (int v = 0; v < 8; ++v) mf[v] = bv;
  }
#pragma unroll
  for (int kc = 0; kc < 13; ++kc)
    mf = WMMA_BF16(load_a_frag(&h1[w][0], 416, kc), load_b_frag(We2, 16, kc, 0), mf);
#pragma unroll
  for (int v = 0; v < 8; ++v) {
    float x = siluf(mf[v]);
    mf[v] = x;
    mst[w][(rbase + v) * 32 + col] = f2bf(x);
    mst[w][(rbase + v) * 32 + 16 + col] = 0;   // zero K-pad 16..31
  }
  // scatter m_i = segment_sum(emask * m, col)
#pragma unroll
  for (int v = 0; v < 8; ++v) {
    int eIdx = rbase + v;
    atomicAdd(&mAgg[(size_t)cn[w][eIdx] * 16 + col], em[w][eIdx] * mf[v]);
  }

  // ---- coors MLP: t = SiLU(m @ Wc1 + bc1) [16,64] ----
  for (int nt = 0; nt < 4; ++nt) {
    int gc = nt * 16 + col;
    float bv = bc1[gc];
    v8f acc;
#pragma unroll
    for (int v = 0; v < 8; ++v) acc[v] = bv;
    acc = WMMA_BF16(load_a_frag(&mst[w][0], 32, 0), load_b_frag(Wc1, 64, 0, nt), acc);
#pragma unroll
    for (int v = 0; v < 8; ++v)
      tst[w][(rbase + v) * 64 + gc] = siluf(acc[v]);
  }
  // w = t @ Wc2 + bc2 ; coors scatter (lanes 0..15 own one edge each)
  if (lane < 16) {
    float acc = bc2[0];
    for (int c2 = 0; c2 < 64; ++c2) acc += tst[w][lane * 64 + c2] * Wc2[c2];
    float s = coors_scale[0] * em[w][lane] * acc * inv;
    int node = cn[w][lane];
    atomicAdd(&coorsNext[(size_t)node * 3 + 0], relx * s);
    atomicAdd(&coorsNext[(size_t)node * 3 + 1], rely * s);
    atomicAdd(&coorsNext[(size_t)node * 3 + 2], relz * s);
  }
}

// ---------------------------------------------------------------------------
// Node update: fn = graphLN(feats); h = SiLU([fn|m_i] @ Wn1 + bn1) @ Wn2 + bn2
// feats_next = feats + h.   K1 = 80 -> pad 96; mid = 128; out = 64.
// ---------------------------------------------------------------------------
__global__ void node_layer_kernel(
    const float* __restrict__ feats, const float* __restrict__ mAgg,
    const int* __restrict__ batch, const float* __restrict__ stats,
    const float* __restrict__ ln_g, const float* __restrict__ ln_b,
    const unsigned short* __restrict__ Wn1, const float* __restrict__ bn1,
    const unsigned short* __restrict__ Wn2, const float* __restrict__ bn2,
    float* __restrict__ featsNext, int N) {
  constexpr int WAVES = 4;
  __shared__ alignas(16) unsigned short a0[WAVES][16 * 96];
  __shared__ alignas(16) unsigned short hs[WAVES][16 * 128];
  int w = threadIdx.x >> 5, lane = threadIdx.x & 31;
  int n0 = (blockIdx.x * WAVES + w) * 16;
  if (n0 >= N) return;
  {
    int L = lane & 15;
    int node = n0 + L;
    int g = batch[node];
    float norm = fmaxf(stats[g * 3 + 0], 1.0f) * (float)HH;
    float mean = stats[g * 3 + 1] / norm;
    float var  = stats[g * 3 + 2] / norm - mean * mean;
    float rstd = rsqrtf(fmaxf(var, 0.0f) + 1e-5f);
    const float* frow = feats + (size_t)node * HH;
    if (lane < 16) {
      for (int c = 0; c < 32; ++c)
        a0[w][L * 96 + c] = f2bf((frow[c] - mean) * rstd * ln_g[c] + ln_b[c]);
    } else {
      for (int c = 32; c < 64; ++c)
        a0[w][L * 96 + c] = f2bf((frow[c] - mean) * rstd * ln_g[c] + ln_b[c]);
      const float* mrow = mAgg + (size_t)node * 16;
      for (int c = 0; c < 16; ++c) a0[w][L * 96 + 64 + c] = f2bf(mrow[c]);
      for (int c = 80; c < 96; ++c) a0[w][L * 96 + c] = 0;
    }
  }
  int col = lane & 15, rbase = (lane & 16) ? 8 : 0;
  __builtin_prefetch(Wn1, 0, 1);
  for (int nt = 0; nt < 8; ++nt) {
    int gc = nt * 16 + col;
    float bv = bn1[gc];
    v8f acc;
#pragma unroll
    for (int v = 0; v < 8; ++v) acc[v] = bv;
#pragma unroll
    for (int kc = 0; kc < 3; ++kc)
      acc = WMMA_BF16(load_a_frag(&a0[w][0], 96, kc), load_b_frag(Wn1, 128, kc, nt), acc);
#pragma unroll
    for (int v = 0; v < 8; ++v)
      hs[w][(rbase + v) * 128 + gc] = f2bf(siluf(acc[v]));
  }
  for (int nt = 0; nt < 4; ++nt) {
    int gc = nt * 16 + col;
    float bv = bn2[gc];
    v8f acc;
#pragma unroll
    for (int v = 0; v < 8; ++v) acc[v] = bv;
#pragma unroll
    for (int kc = 0; kc < 4; ++kc)
      acc = WMMA_BF16(load_a_frag(&hs[w][0], 128, kc), load_b_frag(Wn2, 64, kc, nt), acc);
#pragma unroll
    for (int v = 0; v < 8; ++v) {
      size_t idx = (size_t)(n0 + rbase + v) * HH + gc;
      featsNext[idx] = feats[idx] + acc[v];   // residual
    }
  }
}

__global__ void pack_out_kernel(const float* __restrict__ coors, const float* __restrict__ feats,
                                float* __restrict__ out, int N) {
  int i = blockIdx.x * blockDim.x + threadIdx.x;
  if (i >= N * 67) return;
  int n = i / 67, c = i - n * 67;
  out[i] = (c < 3) ? coors[n * 3 + c] : feats[(size_t)n * HH + (c - 3)];
}

// ---------------------------------------------------------------------------
// Host driver
// ---------------------------------------------------------------------------
static inline size_t alignUp(size_t x) { return (x + 255) & ~(size_t)255; }

extern "C" void kernel_launch(void* const* d_in, const int* in_sizes, int n_in,
                              void* d_out, int out_size, void* d_ws, size_t ws_size,
                              hipStream_t stream) {
  (void)in_sizes; (void)n_in; (void)out_size; (void)ws_size;
  const float* pos_r     = (const float*)d_in[0];
  const float* x_in      = (const float*)d_in[1];
  const float* edge_attr = (const float*)d_in[2];
  const int*   ei0       = (const int*)d_in[3];
  const int*   ei1       = ((const int*)d_in[3]) + EE;
  const int*   batch     = (const int*)d_in[4];
  const float* Wni = (const float*)d_in[5]; const float* bni = (const float*)d_in[6];
  const float* Wei = (const float*)d_in[7]; const float* bei = (const float*)d_in[8];

  // workspace bump allocator
  char* ws = (char*)d_ws; size_t off = 0;
  auto alloc = [&](size_t bytes) { char* p = ws + off; off = alignUp(off + bytes); return p; };
  float* featsA = (float*)alloc((size_t)NN * HH * 4);
  float* featsB = (float*)alloc((size_t)NN * HH * 4);
  float* coorsA = (float*)alloc((size_t)NN * 3 * 4);
  float* coorsB = (float*)alloc((size_t)NN * 3 * 4);
  unsigned short* featsH    = (unsigned short*)alloc((size_t)NN * HH * 2);
  unsigned short* edgeFeatH = (unsigned short*)alloc((size_t)EE * HH * 2);
  float* emask = (float*)alloc((size_t)EE * 4);
  float* mAgg  = (float*)alloc((size_t)NN * 16 * 4);
  float* stats = (float*)alloc(NGRAPH * 3 * 4);
  unsigned short* WniH = (unsigned short*)alloc(32 * 64 * 2);
  unsigned short* WeiH = (unsigned short*)alloc(32 * 64 * 2);
  unsigned short *We1H[3], *We2H[3], *Wc1H[3], *Wn1H[3], *Wn2H[3];
  for (int l = 0; l < 3; ++l) {
    We1H[l] = (unsigned short*)alloc(224 * 416 * 2);
    We2H[l] = (unsigned short*)alloc(416 * 16 * 2);
    Wc1H[l] = (unsigned short*)alloc(32 * 64 * 2);
    Wn1H[l] = (unsigned short*)alloc(96 * 128 * 2);
    Wn2H[l] = (unsigned short*)alloc(128 * 64 * 2);
  }

  auto conv = [&](const float* src, int r, int c, unsigned short* dst, int rp, int cp) {
    int tot = rp * cp;
    convert_pad_kernel<<<(tot + 255) / 256, 256, 0, stream>>>(src, r, c, dst, rp, cp);
  };
  conv(Wni, 32, 64, WniH, 32, 64);
  conv(Wei, 16, 64, WeiH, 32, 64);
  for (int l = 0; l < 3; ++l) {
    const int b = 9 + 15 * l;
    conv((const float*)d_in[b + 0], 203, 406, We1H[l], 224, 416);
    conv((const float*)d_in[b + 2], 406, 16,  We2H[l], 416, 16);
    conv((const float*)d_in[b + 4], 16,  64,  Wc1H[l], 32,  64);
    conv((const float*)d_in[b + 8], 80,  128, Wn1H[l], 96,  128);
    conv((const float*)d_in[b + 10],128, 64,  Wn2H[l], 128, 64);
  }

  emask_kernel<<<(EE + 255) / 256, 256, 0, stream>>>(pos_r, ei0, ei1, emask, EE);

  // node_init: feats = x @ Wni + bni   (f32 out)
  wmma_gemm_kernel<32, false, false><<<(NN / 16 + 3) / 4, 128, 0, stream>>>(
      x_in, NN, 32, WniH, 64, bni, featsA, nullptr);
  // edge_init: edge_feat = edge_attr @ Wei + bei   (bf16 out)
  wmma_gemm_kernel<32, false, true><<<(EE / 16 + 3) / 4, 128, 0, stream>>>(
      edge_attr, EE, 16, WeiH, 64, bei, nullptr, edgeFeatH);

  hipMemcpyAsync(coorsA, pos_r, (size_t)NN * 3 * 4, hipMemcpyDeviceToDevice, stream);

  float* featsCur = featsA; float* featsNxt = featsB;
  float* coorsCur = coorsA; float* coorsNxt = coorsB;
  for (int l = 0; l < 3; ++l) {
    const int b = 9 + 15 * l;
    const float* be1 = (const float*)d_in[b + 1];
    const float* be2 = (const float*)d_in[b + 3];
    const float* bc1 = (const float*)d_in[b + 5];
    const float* Wc2 = (const float*)d_in[b + 6];
    const float* bc2 = (const float*)d_in[b + 7];
    const float* bn1 = (const float*)d_in[b + 9];
    const float* bn2 = (const float*)d_in[b + 11];
    const float* lng = (const float*)d_in[b + 12];
    const float* lnb = (const float*)d_in[b + 13];
    const float* csc = (const float*)d_in[b + 14];

    f32_to_bf16_kernel<<<(NN * HH + 255) / 256, 256, 0, stream>>>(featsCur, featsH, NN * HH);
    hipMemsetAsync(mAgg, 0, (size_t)NN * 16 * 4, stream);
    hipMemsetAsync(stats, 0, NGRAPH * 3 * 4, stream);
    hipMemcpyAsync(coorsNxt, coorsCur, (size_t)NN * 3 * 4, hipMemcpyDeviceToDevice, stream);

    graph_stats_kernel<<<(NN + 255) / 256, 256, 0, stream>>>(featsCur, batch, stats, NN);

    edge_layer_kernel<<<EE / 16 / 2, 64, 0, stream>>>(
        coorsCur, featsH, edgeFeatH, emask, ei0, ei1,
        We1H[l], be1, We2H[l], be2, Wc1H[l], bc1, Wc2, bc2, csc,
        coorsNxt, mAgg, EE);

    node_layer_kernel<<<(NN / 16 + 3) / 4, 128, 0, stream>>>(
        featsCur, mAgg, batch, stats, lng, lnb,
        Wn1H[l], bn1, Wn2H[l], bn2, featsNxt, NN);

    float* t = featsCur; featsCur = featsNxt; featsNxt = t;
    t = coorsCur; coorsCur = coorsNxt; coorsNxt = t;
  }

  pack_out_kernel<<<(NN * 67 + 255) / 256, 256, 0, stream>>>(
      coorsCur, featsCur, (float*)d_out, NN);
}
